// SWindowAttention_34376918237365
// MI455X (gfx1250) — matrix operand order repaired and checked by reading, not compiled, not run on hardware
//
#include <hip/hip_runtime.h>

// ---------------- types -------------------------------------------------
typedef __attribute__((ext_vector_type(2)))  float    v2f;
typedef __attribute__((ext_vector_type(4)))  float    v4f;
typedef __attribute__((ext_vector_type(8)))  float    v8f;
typedef __attribute__((ext_vector_type(8)))  _Float16 v8h;
typedef __attribute__((ext_vector_type(16))) _Float16 v16h;

union H16 { v16h v; v8h h[2]; };

#define DIM    384
#define HEADS  12
#define HD     32
#define NTOK   49
#define NPAD   64
#define TT     32
#define BN     32
#define SCALE  0.17677669529663687f   /* 32^-0.5 */
#define POSMAX 4.0f                   /* level//2 - 1 */
#define NEGMIN (-5.0f)                /* -(level//2) */
#define WSTR   388                    /* LDS W row stride: 388%64=4 -> no bank conflicts */

__device__ __forceinline__ v8f splat8(float v) {
  v8f r;
#pragma unroll
  for (int i = 0; i < 8; ++i) r[i] = v;
  return r;
}

// ---------------- K0a: proj_w fp32 -> f16 -------------------------------
__global__ void cvt_f16_kernel(const float* __restrict__ w,
                               _Float16* __restrict__ w16, int n) {
  int i = blockIdx.x * blockDim.x + threadIdx.x;
  if (i < n) w16[i] = (_Float16)w[i];
}

// ---------------- K0b: zero pad row (A-operand source for pad tokens) ---
__global__ void zero_row_kernel(float* __restrict__ z, int n) {
  int i = blockIdx.x * blockDim.x + threadIdx.x;
  if (i < n) z[i] = 0.0f;
}

// ---------------- K1: fused QKV GEMM (f32 WMMA) + ST-BIF ----------------
// grid (BN=32, 18 col-tiles of 64), block 128 (4 waves).
// The 64x384 fp32 weight tile is t-invariant: staged once into LDS (97 KB,
// padded stride 388 -> conflict-free ds_load_b64 B-fragments). The x (A)
// fragment is software-pipelined one k-step ahead. Neuron state lives in
// VGPRs across the T=32 loop; only exact-f16 cumulative counts are emitted.
__global__ __launch_bounds__(128) void qkv_bif_kernel(
    const float* __restrict__ x, const float* __restrict__ qkv_w,
    const float* __restrict__ qkv_b, const float* __restrict__ zrow,
    _Float16* __restrict__ qa, _Float16* __restrict__ ka,
    _Float16* __restrict__ vaT) {
  __shared__ __align__(16) float lds_w[64 * WSTR];   // 99328 B

  const int b  = blockIdx.x;
  const int ct = blockIdx.y;
  const int wv = threadIdx.x >> 5;
  const int ln = threadIdx.x & 31;
  const int lo = ln & 15;
  const int hi = ln >> 4;

  const int m    = (wv << 4) + lo;   // A row (token) this lane feeds
  const int col0 = ct * 64;
  const int sel  = ct / 6;           // 0=q 1=k 2=v

  // ---- stage W tile (64 cols x 384 k) into LDS, v4f granules -----------
  for (int idx = threadIdx.x; idx < 64 * (DIM / 4); idx += 128) {
    const int c  = idx / (DIM / 4);
    const int k4 = (idx % (DIM / 4)) * 4;
    v4f wv4 = *(const v4f*)(qkv_w + (size_t)(col0 + c) * DIM + k4);
    *(v4f*)&lds_w[c * WSTR + k4] = wv4;
  }

  // bias per owned column (t-invariant)
  float bb[4];
#pragma unroll
  for (int nt = 0; nt < 4; ++nt) bb[nt] = qkv_b[col0 + nt * 16 + lo];

  float vmem[4][8], qc[4][8];
#pragma unroll
  for (int nt = 0; nt < 4; ++nt)
#pragma unroll
    for (int r = 0; r < 8; ++r) { vmem[nt][r] = 0.5f; qc[nt][r] = 0.0f; }

  __syncthreads();

  for (int t = 0; t < TT; ++t) {
    const int r = t * BN + b;
    // per-lane base: valid token row of x, or the shared zero row
    const float* xrow =
        (m < NTOK) ? (x + (size_t)(r * NTOK + m) * DIM) : zrow;

    v8f acc[4];
#pragma unroll
    for (int nt = 0; nt < 4; ++nt) acc[nt] = splat8(bb[nt]);

    v2f a = *(const v2f*)(xrow + 2 * hi);          // k-step 0 A fragment
    for (int k0 = 0; k0 < DIM - 4; k0 += 4) {
      v2f aN = *(const v2f*)(xrow + k0 + 4 + 2 * hi);  // next step, in flight
#pragma unroll
      for (int nt = 0; nt < 4; ++nt) {
        v2f bm = *(const v2f*)&lds_w[(nt * 16 + lo) * WSTR + k0 + 2 * hi];
        acc[nt] = __builtin_amdgcn_wmma_f32_16x16x4_f32(
            false, a, false, bm, (short)0, acc[nt], false, false);
      }
      a = aN;
    }
#pragma unroll
    for (int nt = 0; nt < 4; ++nt) {               // final k-step (DIM-4)
      v2f bm = *(const v2f*)&lds_w[(nt * 16 + lo) * WSTR + (DIM - 4) + 2 * hi];
      acc[nt] = __builtin_amdgcn_wmma_f32_16x16x4_f32(
          false, a, false, bm, (short)0, acc[nt], false, false);
    }

    // ST-BIF epilogue; emit cumulative count q (thr=1) as exact f16
#pragma unroll
    for (int nt = 0; nt < 4; ++nt) {
      const int col = col0 + nt * 16 + lo;
      const int cc  = col - sel * DIM;
      const int h   = cc >> 5;
      const int d   = cc & 31;
      const size_t base = (size_t)r * HEADS + h;
      _Float16* dst;                                // dst[tok * stride]
      int stride;
      if (sel == 0)      { dst = qa  + (base * NPAD) * HD + d; stride = HD; }
      else if (sel == 1) { dst = ka  + (base * NPAD) * HD + d; stride = HD; }
      else               { dst = vaT + (base * HD + d) * NPAD; stride = 1;  }
#pragma unroll
      for (int rr = 0; rr < 8; ++rr) {
        const int tok = (wv << 4) + rr + (hi << 3);
        float v = vmem[nt][rr] + acc[nt][rr];
        float q = qc[nt][rr];
        float s = 0.0f;
        if (v >= 1.0f && q < POSMAX) s = 1.0f;
        else if (v < 0.0f && q > NEGMIN) s = -1.0f;
        v -= s; q += s;
        vmem[nt][rr] = v; qc[nt][rr] = q;
        dst[tok * stride] = (_Float16)((tok < NTOK) ? q : 0.0f);
      }
    }
  }
}

// ---------------- K2: attention core ------------------------------------
// grid (BN=32, HEADS=12), block 128. Per t:
//   S(t) = qa*ka^T (f16 WMMA, K=32, exact);  attn_t = SCALE*(S-Sprev)+bias+mask
//   IF neuron on attn_t/49 (state in VGPRs) -> a_a -> LDS (layout swap)
//   O(t) = a_a * v_a (f16 WMMA, K=64 padded);  out_t = O - Oprev
__global__ __launch_bounds__(128) void attn_core_kernel(
    const _Float16* __restrict__ qa, const _Float16* __restrict__ ka,
    const _Float16* __restrict__ vaT,
    const float* __restrict__ mask, const float* __restrict__ bias_table,
    const int* __restrict__ rel_index, float* __restrict__ outb) {
  __shared__ __align__(32) _Float16 lds_aa[NPAD * NPAD];

  const int b  = blockIdx.x;
  const int h  = blockIdx.y;
  const int wv = threadIdx.x >> 5;
  const int ln = threadIdx.x & 31;
  const int lo = ln & 15;
  const int hi = ln >> 4;
  const int mrow = (wv << 4) + lo;

  // per-element mask / (bias/STEP), zero on pads
  float maskE[4][8], biasE[4][8];
#pragma unroll
  for (int nt = 0; nt < 4; ++nt) {
    const int j = nt * 16 + lo;
#pragma unroll
    for (int rr = 0; rr < 8; ++rr) {
      const int i = (wv << 4) + rr + (hi << 3);
      float mv = 0.0f, bv = 0.0f;
      if (i < NTOK && j < NTOK) {
        mv = mask[(size_t)b * NTOK * NTOK + i * NTOK + j];
        bv = bias_table[rel_index[i * NTOK + j] * HEADS + h] * 0.25f;
      }
      maskE[nt][rr] = mv; biasE[nt][rr] = bv;
    }
  }

  v8f Sprev[4], Oprev[2];
  float vm[4][8], qc[4][8];
#pragma unroll
  for (int nt = 0; nt < 4; ++nt) {
    Sprev[nt] = splat8(0.0f);
#pragma unroll
    for (int rr = 0; rr < 8; ++rr) { vm[nt][rr] = 0.5f; qc[nt][rr] = 0.0f; }
  }
  Oprev[0] = splat8(0.0f); Oprev[1] = splat8(0.0f);
  const v8f zero8 = splat8(0.0f);

  for (int t = 0; t < TT; ++t) {
    const size_t r  = (size_t)(t * BN + b);
    const size_t bq = r * HEADS + h;

    // A fragment: qa row mrow (16x32 f16 layout: contiguous 16 at 16*hi)
    H16 aF;
    const _Float16* qb = qa + (bq * NPAD + mrow) * HD;
    aF.v = *(const v16h*)(qb + 16 * hi);

    v8f Scur[4];
#pragma unroll
    for (int nt = 0; nt < 4; ++nt) {
      const _Float16* kb = ka + (bq * NPAD + nt * 16 + lo) * HD;
      v16h bF = *(const v16h*)(kb + 16 * hi);  // B 32x16: contiguous 16 at 16*hi
      Scur[nt] = __builtin_amdgcn_wmma_f32_16x16x32_f16(
          false, aF.v, false, bF, (short)0, zero8, false, false);
    }

    // IF neuron on attn/N; write cumulative a_a to LDS (pads exact zero)
#pragma unroll
    for (int nt = 0; nt < 4; ++nt) {
      const int j = nt * 16 + lo;
#pragma unroll
      for (int rr = 0; rr < 8; ++rr) {
        const int i = (wv << 4) + rr + (hi << 3);
        float xin = (SCALE * (Scur[nt][rr] - Sprev[nt][rr]) + maskE[nt][rr] +
                     (t < 4 ? biasE[nt][rr] : 0.0f)) * (1.0f / NTOK);
        float v = vm[nt][rr] + xin;
        float q = qc[nt][rr];
        float s = 0.0f;
        if (v >= 1.0f && q < POSMAX) s = 1.0f;
        else if (v < 0.0f && q > NEGMIN) s = -1.0f;
        v -= s; q += s;
        vm[nt][rr] = v; qc[nt][rr] = q;
        lds_aa[i * NPAD + j] = (_Float16)q;
      }
      Sprev[nt] = Scur[nt];
    }
    __syncthreads();

    // O = a_a(64x64) * v_a(64x32), K padded with exact zeros
    v8f Ocur[2]; Ocur[0] = splat8(0.0f); Ocur[1] = splat8(0.0f);
#pragma unroll
    for (int k0 = 0; k0 < NPAD; k0 += 32) {
      H16 aA;
      aA.v = *(const v16h*)&lds_aa[mrow * NPAD + k0 + 16 * hi];
#pragma unroll
      for (int nt2 = 0; nt2 < 2; ++nt2) {
        const _Float16* vb = vaT + (bq * HD + nt2 * 16 + lo) * NPAD;
        v16h bF = *(const v16h*)(vb + k0 + 16 * hi);
        Ocur[nt2] = __builtin_amdgcn_wmma_f32_16x16x32_f16(
            false, aA.v, false, bF, (short)0, Ocur[nt2], false, false);
      }
    }

    // out_t = O - Oprev  -> [r, tok, h*32+d] fp32
#pragma unroll
    for (int nt2 = 0; nt2 < 2; ++nt2) {
      const int c = h * HD + nt2 * 16 + lo;
#pragma unroll
      for (int rr = 0; rr < 8; ++rr) {
        const int i = (wv << 4) + rr + (hi << 3);
        if (i < NTOK)
          outb[(r * NTOK + i) * DIM + c] = Ocur[nt2][rr] - Oprev[nt2][rr];
      }
      Oprev[nt2] = Ocur[nt2];
    }
    __syncthreads();
  }
}

// ---------------- K3: output ST-BIF (spikes in {-1,0,1}) ----------------
__global__ void out_bif_kernel(const float* __restrict__ outb,
                               _Float16* __restrict__ outs) {
  const int per = BN * NTOK * DIM;
  int e = blockIdx.x * blockDim.x + threadIdx.x;
  if (e >= per) return;
  const int b   = e / (NTOK * DIM);
  const int rem = e % (NTOK * DIM);
  float v = 0.5f, q = 0.0f;
  for (int t = 0; t < TT; ++t) {
    size_t idx = (size_t)(t * BN + b) * NTOK * DIM + rem;
    v += outb[idx];
    float s = 0.0f;
    if (v >= 1.0f && q < POSMAX) s = 1.0f;
    else if (v < 0.0f && q > NEGMIN) s = -1.0f;
    v -= s; q += s;
    outs[idx] = (_Float16)s;   // spike * thr, exact
  }
}

// ---------------- K4: projection GEMM (f16 WMMA) ------------------------
// grid (784 M-tiles, 6 N-tiles), block 128. 50176 rows (64 | 50176).
__global__ __launch_bounds__(128) void proj_kernel(
    const _Float16* __restrict__ outs, const _Float16* __restrict__ pw16,
    const float* __restrict__ proj_b, float* __restrict__ out) {
  const int bm = blockIdx.x, bn = blockIdx.y;
  const int wv = threadIdx.x >> 5;
  const int ln = threadIdx.x & 31;
  const int lo = ln & 15;
  const int hi = ln >> 4;
  const int row = bm * 64 + (wv << 4) + lo;

  v8f acc[4];
#pragma unroll
  for (int nt = 0; nt < 4; ++nt) acc[nt] = splat8(0.0f);

  const _Float16* ab = outs + (size_t)row * DIM;
#pragma unroll
  for (int k0 = 0; k0 < DIM; k0 += 32) {
    H16 aF;
    aF.v = *(const v16h*)(ab + k0 + 16 * hi);
#pragma unroll
    for (int nt = 0; nt < 4; ++nt) {
      const int col = bn * 64 + nt * 16 + lo;
      v16h bF = *(const v16h*)(pw16 + (size_t)col * DIM + k0 + 16 * hi);
      acc[nt] = __builtin_amdgcn_wmma_f32_16x16x32_f16(
          false, aF.v, false, bF, (short)0, acc[nt], false, false);
    }
  }
#pragma unroll
  for (int nt = 0; nt < 4; ++nt) {
    const int col = bn * 64 + nt * 16 + lo;
    const float pb = proj_b[col];
#pragma unroll
    for (int rr = 0; rr < 8; ++rr) {
      const int grow = bm * 64 + (wv << 4) + rr + (hi << 3);
      out[(size_t)grow * DIM + col] = acc[nt][rr] + pb;
    }
  }
}

// ---------------- launcher ----------------------------------------------
extern "C" void kernel_launch(void* const* d_in, const int* in_sizes, int n_in,
                              void* d_out, int out_size, void* d_ws,
                              size_t ws_size, hipStream_t stream) {
  (void)in_sizes; (void)n_in; (void)out_size; (void)ws_size;
  const float* x          = (const float*)d_in[0];
  const float* mask       = (const float*)d_in[1];
  const float* qkv_w      = (const float*)d_in[2];
  const float* qkv_b      = (const float*)d_in[3];
  const float* proj_w     = (const float*)d_in[4];
  const float* proj_b     = (const float*)d_in[5];
  const float* bias_table = (const float*)d_in[6];
  const int*   rel_index  = (const int*)d_in[7];

  char* ws = (char*)d_ws;
  _Float16* qa   = (_Float16*)(ws);                 //  50331648 B
  _Float16* ka   = (_Float16*)(ws + 50331648u);     //  50331648 B
  _Float16* vaT  = (_Float16*)(ws + 100663296u);    //  50331648 B
  float*    attn = (float*)  (ws + 150994944u);     //  77070336 B
  _Float16* outs = (_Float16*)(ws + 228065280u);    //  38535168 B
  _Float16* pw16 = (_Float16*)(ws + 266600448u);    //    294912 B
  float*    zrow = (float*)  (ws + 266895360u);     //      1536 B
  float* out = (float*)d_out;

  zero_row_kernel<<<2, 256, 0, stream>>>(zrow, DIM);
  cvt_f16_kernel<<<(DIM * DIM + 255) / 256, 256, 0, stream>>>(proj_w, pw16,
                                                              DIM * DIM);
  qkv_bif_kernel<<<dim3(BN, 18), 128, 0, stream>>>(x, qkv_w, qkv_b, zrow,
                                                   qa, ka, vaT);
  attn_core_kernel<<<dim3(BN, HEADS), 128, 0, stream>>>(qa, ka, vaT, mask,
                                                        bias_table, rel_index,
                                                        attn);
  out_bif_kernel<<<(BN * NTOK * DIM + 255) / 256, 256, 0, stream>>>(attn, outs);
  proj_kernel<<<dim3(784, 6), 128, 0, stream>>>(outs, pw16, proj_b, out);
}
